// GFLayerNSF_14998025797943
// MI455X (gfx1250) — matrix operand order repaired and checked by reading, not compiled, not run on hardware
//
#include <hip/hip_runtime.h>

typedef __attribute__((ext_vector_type(16))) _Float16 v16h;
typedef __attribute__((ext_vector_type(8)))  _Float16 v8h;
typedef __attribute__((ext_vector_type(8)))  float    v8f;

#define NBATCH 32
#define NSEQ   1024
#define E_     32
#define E2_    16
#define HID_   64
#define DVDIM  224          // (3*5-1)*16
#define NTOK   (NBATCH*NSEQ)
#define TBND   3.0f

// ---------------------------------------------------------------- WMMA helpers

__device__ __forceinline__ v8f wmma_f16(v16h a, v16h b, v8f c) {
  return __builtin_amdgcn_wmma_f32_16x16x32_f16(
      /*neg_a=*/false, a, /*neg_b=*/false, b,
      /*c_mod=*/(short)0, c, /*reuse_a=*/false, /*reuse_b=*/false);
}

// Load 16 contiguous halves as two 16B vector loads (16B alignment required).
__device__ __forceinline__ v16h load16(const _Float16* p) {
  v8h lo = *(const v8h*)p;
  v8h hi = *(const v8h*)(p + 8);
  return __builtin_shufflevector(lo, hi, 0, 1, 2, 3, 4, 5, 6, 7,
                                 8, 9, 10, 11, 12, 13, 14, 15);
}

// A fragment (16x32 f16, M x K) from a row-major source.
// lane holds row = lane&15; halves are two contiguous 16B chunks:
//   K = hlf*16 .. +7   and   K = hlf*16 + 8 .. +7
__device__ __forceinline__ v16h load_a_frag_h(const _Float16* src, int ld) {
  const int lane = threadIdx.x & 31;
  const int row  = lane & 15;
  const int hlf  = lane >> 4;
  const _Float16* p = src + row * ld + hlf * 16;
  v8h lo = *(const v8h*)p;
  v8h hi = *(const v8h*)(p + 8);
  return __builtin_shufflevector(lo, hi, 0, 1, 2, 3, 4, 5, 6, 7,
                                 8, 9, 10, 11, 12, 13, 14, 15);
}

// B fragments are staged in LDS pre-swizzled to per-lane order:
//   tile t (32 K-values x 16 N): slot [t*512 + lane*16 + e] holds
//   B[k][n] with k = (lane>>4)*16 + e, n = lane&15.
// A fragment load is then 32 contiguous bytes -> 2x ds_load_b128.

// ---------------------------------------------------------------- GEMM kernel
// out[M,Nn] = act(A[M,K] @ W[K,Nn] + bias)   (optionally + residual A)
// block = 256 threads = 8 waves; block tile = 128 rows x 16 cols.
__global__ void gemm_wmma(const float* __restrict__ A, const float* __restrict__ W,
                          const float* __restrict__ bias,
                          float* __restrict__ outF, _Float16* __restrict__ outH,
                          int K, int Kpad, int Nn, int act, int resid) {
  extern __shared__ char smem[];
  _Float16* As  = (_Float16*)smem;                       // 128 x Kpad, row-major
  _Float16* Wsw = As + 128 * Kpad;                       // (Kpad/32) x 512 swizzled
  float*    bs  = (float*)(Wsw + (Kpad >> 5) * 512);     // 16

  const int tid  = threadIdx.x;
  const int wave = tid >> 5;
  const int lane = tid & 31;
  const int m0   = blockIdx.x * 128;
  const int n0   = blockIdx.y * 16;
  const int ntiles = Kpad >> 5;

  // A block: 16 threads per row, consecutive columns (coalesced 64B).
  for (int r = tid >> 4; r < 128; r += 16) {
    const float* Ar = A + (long)(m0 + r) * K;
    for (int c = tid & 15; c < Kpad; c += 16)
      As[r * Kpad + c] = (c < K) ? (_Float16)Ar[c] : (_Float16)0.f;
  }
  // W column slice, swizzled into B-fragment order.
  for (int i = tid; i < ntiles * 512; i += 256) {
    int t = i >> 9, rem = i & 511, ln = rem >> 4, e = rem & 15;
    int k = t * 32 + ((ln >> 4) << 4) + e;
    int n = ln & 15;
    Wsw[i] = (k < K) ? (_Float16)W[(long)k * Nn + n0 + n] : (_Float16)0.f;
  }
  if (tid < 16) bs[tid] = bias[n0 + tid];
  __syncthreads();

  v8f c = {};
  for (int t = 0; t < ntiles; ++t) {
    v16h af = load_a_frag_h(&As[wave * 16 * Kpad + t * 32], Kpad);
    v16h bf = load16(&Wsw[t * 512 + lane * 16]);
    c = wmma_f16(af, bf, c);
  }

  const int hlf = lane >> 4, n = lane & 15;
#pragma unroll
  for (int r = 0; r < 8; ++r) {
    int  m   = (hlf << 3) + r;
    long row = (long)m0 + wave * 16 + m;
    float v = c[r] + bs[n];
    if (resid)      v = A[row * K + n0 + n] + fmaxf(v, 0.f);  // o + relu(o@W+b)
    else if (act)   v = fmaxf(v, 0.f);
    if (outH) outH[row * Nn + n0 + n] = (_Float16)v;
    else      outF[row * Nn + n0 + n] = v;
  }
}

// ------------------------------------------------------- flash-attention kernel
// o = q + softmax(q k^T / sqrt(DV)) v      per batch, N=1024, DV=224, f16 in.
// block = 256 threads (8 waves x 16 queries); 32-key tiles staged in LDS,
// K pre-transposed and V pre-swizzled into B-fragment order.
__global__ void attn_kernel(const _Float16* __restrict__ Q,
                            const _Float16* __restrict__ Km,
                            const _Float16* __restrict__ Vm,
                            float* __restrict__ O) {
  __shared__ alignas(32) _Float16 Ksw[14 * 512];  // [ntile*7+dt][lane][16]
  __shared__ alignas(32) _Float16 Vsw[14 * 512];  // [dtile][lane][16]
  __shared__ alignas(32) _Float16 Ps[8 * 16 * 32];

  const int tid  = threadIdx.x;
  const int wave = tid >> 5;
  const int lane = tid & 31;
  const int hlf  = lane >> 4;
  const int ncol = lane & 15;
  const int b    = blockIdx.y;
  const long bTok  = (long)b * NSEQ;
  const int  qtok0 = blockIdx.x * 128 + wave * 16;
  const _Float16* Qrow = Q + (bTok + qtok0) * DVDIM;

  v16h qf[7];
#pragma unroll
  for (int dt = 0; dt < 7; ++dt) qf[dt] = load_a_frag_h(Qrow + dt * 32, DVDIM);

  v8f   oacc[14] = {};
  float rowm[8], rowl[8];
#pragma unroll
  for (int r = 0; r < 8; ++r) { rowm[r] = -1e30f; rowl[r] = 0.f; }

  const float scale = 0.066815309f;  // 1/sqrt(224)

  for (int kb = 0; kb < NSEQ; kb += 32) {
    __syncthreads();
    const _Float16* Kg = Km + (bTok + kb) * DVDIM;
    const _Float16* Vg = Vm + (bTok + kb) * DVDIM;
    // Swizzle-stage K (transposed into S B-layout) and V (into PV B-layout).
    for (int i = tid; i < 14 * 512; i += 256) {
      int tile = i >> 9;            // 0..13
      int ln   = (i >> 4) & 31;
      int e    = i & 15;
      int nt   = (tile >= 7) ? 1 : 0;
      int dt   = tile - nt * 7;
      int key  = nt * 16 + (ln & 15);
      int d    = dt * 32 + ((ln >> 4) << 4) + e;
      Ksw[i] = Kg[key * DVDIM + d];
      int kk = ((ln >> 4) << 4) + e;
      Vsw[i] = Vg[kk * DVDIM + tile * 16 + (ln & 15)];
    }
    __syncthreads();
    if (kb + 32 < NSEQ) __builtin_prefetch(Km + (bTok + kb + 32) * DVDIM, 0, 0);

    // S (16 queries x 32 keys) = sum over 7 d-tiles
    v8f s0 = {}, s1 = {};
#pragma unroll
    for (int dt = 0; dt < 7; ++dt) {
      s0 = wmma_f16(qf[dt], load16(&Ksw[dt * 512 + lane * 16]), s0);
      s1 = wmma_f16(qf[dt], load16(&Ksw[(7 + dt) * 512 + lane * 16]), s1);
    }

    // online softmax; row stats per C-row r (M = r + 8*half)
    _Float16* Pw = &Ps[wave * 16 * 32];
#pragma unroll
    for (int r = 0; r < 8; ++r) {
      float a0 = s0[r] * scale, a1 = s1[r] * scale;
      float mx = fmaxf(a0, a1);
      mx = fmaxf(mx, __shfl_xor(mx, 1, 32));
      mx = fmaxf(mx, __shfl_xor(mx, 2, 32));
      mx = fmaxf(mx, __shfl_xor(mx, 4, 32));
      mx = fmaxf(mx, __shfl_xor(mx, 8, 32));
      float nm = fmaxf(rowm[r], mx);
      float sc = __expf(rowm[r] - nm);
      rowm[r] = nm;
      float p0 = __expf(a0 - nm), p1 = __expf(a1 - nm);
      float rs = p0 + p1;
      rs += __shfl_xor(rs, 1, 32);
      rs += __shfl_xor(rs, 2, 32);
      rs += __shfl_xor(rs, 4, 32);
      rs += __shfl_xor(rs, 8, 32);
      rowl[r] = rowl[r] * sc + rs;
#pragma unroll
      for (int t = 0; t < 14; ++t) oacc[t][r] *= sc;
      int m = r + (hlf << 3);
      Pw[m * 32 + ncol]      = (_Float16)p0;   // C-layout -> row-major P tile
      Pw[m * 32 + 16 + ncol] = (_Float16)p1;
    }
    asm volatile("s_wait_dscnt 0" ::: "memory");  // LDS round-trip (same wave)

    v16h pf = load_a_frag_h(Pw, 32);              // P as A fragment (16x32)
#pragma unroll
    for (int t = 0; t < 14; ++t)
      oacc[t] = wmma_f16(pf, load16(&Vsw[t * 512 + lane * 16]), oacc[t]);
  }

#pragma unroll
  for (int t = 0; t < 14; ++t) {
#pragma unroll
    for (int r = 0; r < 8; ++r) {
      int  m   = r + (hlf << 3);
      long row = bTok + qtok0 + m;
      int  d   = t * 16 + ncol;
      float qv = (float)Q[row * DVDIM + d];
      O[row * DVDIM + d] = qv + oacc[t][r] / rowl[r];
    }
  }
}

// ----------------------------------------------------------- small VALU kernels

__global__ void conv_split(const float* __restrict__ x, const float* __restrict__ Wc,
                           float* __restrict__ lower, float* __restrict__ upper) {
  __shared__ float Wcs[E_ * E_];
  for (int i = threadIdx.x; i < E_ * E_; i += 256) Wcs[i] = Wc[i];
  __syncthreads();
  long g = (long)blockIdx.x * 256 + threadIdx.x;
  long tok = g >> 5;
  int  c   = (int)(g & 31);
  const float* xr = x + tok * E_;
  float acc = 0.f;
#pragma unroll
  for (int e = 0; e < E_; ++e) acc += xr[e] * Wcs[e * E_ + c];
  if (c < E2_) lower[tok * E2_ + c] = acc;
  else         upper[tok * E2_ + c - E2_] = acc;
}

// LU w/ partial pivoting for slogdet(Wc); init logdet[b] = N * logabsdet.
__global__ void slogdet_init(const float* __restrict__ Wc, float* __restrict__ logdet) {
  __shared__ float Mx[E_ * E_];
  __shared__ float resv;
  int t = threadIdx.x;
  for (int i = t; i < E_ * E_; i += blockDim.x) Mx[i] = Wc[i];
  __syncthreads();
  if (t == 0) {
    float s = 0.f;
    for (int c = 0; c < E_; ++c) {
      int p = c; float mv = fabsf(Mx[c * E_ + c]);
      for (int r = c + 1; r < E_; ++r) {
        float v = fabsf(Mx[r * E_ + c]);
        if (v > mv) { mv = v; p = r; }
      }
      if (p != c)
        for (int j = c; j < E_; ++j) {
          float tv = Mx[c * E_ + j]; Mx[c * E_ + j] = Mx[p * E_ + j]; Mx[p * E_ + j] = tv;
        }
      float piv = Mx[c * E_ + c];
      s += logf(fabsf(piv));
      float inv = 1.f / piv;
      for (int r = c + 1; r < E_; ++r) {
        float f = Mx[r * E_ + c] * inv;
        for (int j = c + 1; j < E_; ++j) Mx[r * E_ + j] -= f * Mx[c * E_ + j];
      }
    }
    resv = s * (float)NSEQ;
  }
  __syncthreads();
  if (t < NBATCH) logdet[t] = resv;
}

__device__ __forceinline__ float softplus_f(float x) {
  return (x > 20.f) ? x : log1pf(__expf(x));
}
__device__ __forceinline__ void softmax5(float* v) {
  float m = v[0];
#pragma unroll
  for (int i = 1; i < 5; ++i) m = fmaxf(m, v[i]);
  float s = 0.f;
#pragma unroll
  for (int i = 0; i < 5; ++i) { v[i] = __expf(v[i] - m); s += v[i]; }
  float inv = 1.f / s;
#pragma unroll
  for (int i = 0; i < 5; ++i) v[i] *= inv;
}

// rational-quadratic spline, one thread per (token, channel); K=5 bins.
__global__ void spline_rqs(const float* __restrict__ u, const float* __restrict__ raw,
                           float* __restrict__ ynew, float* __restrict__ logdet) {
  __shared__ float red[256];
  long g   = (long)blockIdx.x * 256 + threadIdx.x;
  long tok = g >> 4;
  int  j   = (int)(g & 15);
  int  b   = (int)(tok >> 10);
  const float* rp = raw + tok * DVDIM + j * 14;

  float w5[5], h5[5], d6[6];
#pragma unroll
  for (int i = 0; i < 5; ++i) w5[i] = rp[i];
#pragma unroll
  for (int i = 0; i < 5; ++i) h5[i] = rp[5 + i];
  const float DC = logf(__expf(1.f - 0.001f) - 1.f);  // DCONST
  d6[0] = DC; d6[5] = DC;
#pragma unroll
  for (int i = 0; i < 4; ++i) d6[1 + i] = softplus_f(rp[10 + i]);

  softmax5(w5); softmax5(h5);                    // coupling-level softmax
#pragma unroll
  for (int i = 0; i < 5; ++i) { w5[i] *= 2.f * TBND; h5[i] *= 2.f * TBND; }
  softmax5(w5); softmax5(h5);                    // rqs-level softmax

  float widths[5], cw[6], heights[5], ch[6];
  float acc = 0.f; cw[0] = -TBND;
#pragma unroll
  for (int i = 0; i < 5; ++i) {
    acc += 0.001f + (1.f - 0.005f) * w5[i];
    cw[i + 1] = 2.f * TBND * acc - TBND;
  }
  cw[5] = TBND;
#pragma unroll
  for (int i = 0; i < 5; ++i) widths[i] = cw[i + 1] - cw[i];
  acc = 0.f; ch[0] = -TBND;
#pragma unroll
  for (int i = 0; i < 5; ++i) {
    acc += 0.001f + (1.f - 0.005f) * h5[i];
    ch[i + 1] = 2.f * TBND * acc - TBND;
  }
  ch[5] = TBND;
#pragma unroll
  for (int i = 0; i < 5; ++i) heights[i] = ch[i + 1] - ch[i];
  float der[6];
#pragma unroll
  for (int i = 0; i < 6; ++i) der[i] = 0.001f + softplus_f(d6[i]);

  float xv = u[g];
  bool inside = (xv >= -TBND) && (xv <= TBND);
  float xs = inside ? xv : 0.f;
  int cnt = 0;
#pragma unroll
  for (int i = 0; i < 6; ++i) {
    float edge = cw[i] + ((i == 5) ? 1e-6f : 0.f);
    cnt += (xs >= edge) ? 1 : 0;
  }
  int idx = cnt - 1; idx = idx < 0 ? 0 : (idx > 4 ? 4 : idx);
  float in_cw = cw[idx], in_w = widths[idx];
  float in_ch = ch[idx], in_h = heights[idx];
  float delta = in_h / in_w;
  float d0 = der[idx], d1 = der[idx + 1];
  float th   = (xs - in_cw) / in_w;
  float th1m = th * (1.f - th);
  float num  = in_h * (delta * th * th + d0 * th1m);
  float den  = delta + (d0 + d1 - 2.f * delta) * th1m;
  float y    = in_ch + num / den;
  float omt  = 1.f - th;
  float dnum = delta * delta * (d1 * th * th + 2.f * delta * th1m + d0 * omt * omt);
  float ld   = logf(dnum) - 2.f * logf(den);

  ynew[g] = inside ? y : xv;
  red[threadIdx.x] = inside ? ld : 0.f;
  __syncthreads();
  for (int s = 128; s > 0; s >>= 1) {
    if (threadIdx.x < s) red[threadIdx.x] += red[threadIdx.x + s];
    __syncthreads();
  }
  if (threadIdx.x == 0) atomicAdd(&logdet[b], red[0]);
}

__global__ void concat_out(const float* __restrict__ lower, const float* __restrict__ upper,
                           float* __restrict__ out) {
  long g = (long)blockIdx.x * 256 + threadIdx.x;
  long tok = g >> 5;
  int  c   = (int)(g & 31);
  out[g] = (c < E2_) ? lower[tok * E2_ + c] : upper[tok * E2_ + c - E2_];
}

// ---------------------------------------------------------------- host launch

struct BP {
  const float *l1w, *l1b, *l2w, *l2b, *l3w, *l3b;
  const float *qw, *qb, *kw, *kb, *vw, *vb, *ow, *ob;
};

extern "C" void kernel_launch(void* const* d_in, const int* in_sizes, int n_in,
                              void* d_out, int out_size, void* d_ws, size_t ws_size,
                              hipStream_t stream) {
  (void)out_size; (void)ws_size;
  auto F = [&](int i) { return (const float*)d_in[i]; };

  // Resolve flattening order from in_sizes: x is the 1,048,576-element array.
  bool x_first = (n_in > 0 && in_sizes[0] == NTOK * E_);
  int  convIdx = x_first ? 1 : 0;
  int  base0   = x_first ? 2 : 1;
  int  xIdx    = x_first ? 0 : (n_in - 1);
  bool sortedLin = !(x_first && n_in > 2 && in_sizes[2] == E2_ * HID_);  // l1.w first?

  const float* xin   = F(xIdx);
  const float* convW = F(convIdx);

  BP blk[2];
  for (int f = 0; f < 2; ++f) {
    int base = base0 + f * 14;
    if (!sortedLin) {  // insertion order: l1..l3, q,k,v,o; each (w,b)
      blk[f].l1w = F(base + 0);  blk[f].l1b = F(base + 1);
      blk[f].l2w = F(base + 2);  blk[f].l2b = F(base + 3);
      blk[f].l3w = F(base + 4);  blk[f].l3b = F(base + 5);
      blk[f].qw  = F(base + 6);  blk[f].qb  = F(base + 7);
      blk[f].kw  = F(base + 8);  blk[f].kb  = F(base + 9);
      blk[f].vw  = F(base + 10); blk[f].vb  = F(base + 11);
      blk[f].ow  = F(base + 12); blk[f].ob  = F(base + 13);
    } else {           // sorted pytree: mlp(l1..l3), sab(k,o,q,v); each (b,w)
      blk[f].l1b = F(base + 0);  blk[f].l1w = F(base + 1);
      blk[f].l2b = F(base + 2);  blk[f].l2w = F(base + 3);
      blk[f].l3b = F(base + 4);  blk[f].l3w = F(base + 5);
      blk[f].kb  = F(base + 6);  blk[f].kw  = F(base + 7);
      blk[f].ob  = F(base + 8);  blk[f].ow  = F(base + 9);
      blk[f].qb  = F(base + 10); blk[f].qw  = F(base + 11);
      blk[f].vb  = F(base + 12); blk[f].vw  = F(base + 13);
    }
  }

  // Workspace carve-up (stream-ordered reuse is sequential, no races).
  char*  ws  = (char*)d_ws;
  size_t off = 0;
  auto alloc = [&](size_t bytes) {
    void* p = ws + off;
    off = (off + bytes + 255) & ~(size_t)255;
    return p;
  };
  float*    lower  = (float*)   alloc((size_t)NTOK * E2_  * 4);
  float*    upper  = (float*)   alloc((size_t)NTOK * E2_  * 4);
  float*    hA     = (float*)   alloc((size_t)NTOK * HID_ * 4);
  float*    hB     = (float*)   alloc((size_t)NTOK * HID_ * 4);
  _Float16* qh     = (_Float16*)alloc((size_t)NTOK * DVDIM * 2);
  _Float16* kh     = (_Float16*)alloc((size_t)NTOK * DVDIM * 2);
  _Float16* vh     = (_Float16*)alloc((size_t)NTOK * DVDIM * 2);
  float*    oattn  = (float*)   alloc((size_t)NTOK * DVDIM * 4);
  float*    sabout = (float*)   alloc((size_t)NTOK * DVDIM * 4);

  float* outT  = (float*)d_out;
  float* ldout = outT + (size_t)NTOK * E_;

  auto gemm = [&](const float* A, const float* W, const float* bias,
                  float* oF, _Float16* oH, int K, int Kpad, int Nn, int act, int resid) {
    dim3 grid(NTOK / 128, Nn / 16);
    size_t sh = (size_t)(128 * Kpad + Kpad * 16) * sizeof(_Float16) + 16 * sizeof(float);
    gemm_wmma<<<grid, dim3(256), sh, stream>>>(A, W, bias, oF, oH, K, Kpad, Nn, act, resid);
  };

  auto coupling = [&](const float* cond, float* ubuf, const BP& p) {
    gemm(cond, p.l1w, p.l1b, hA, nullptr, E2_, 32, HID_, 1, 0);
    gemm(hA,   p.l2w, p.l2b, hB, nullptr, HID_, HID_, HID_, 1, 0);
    gemm(hB,   p.l3w, p.l3b, hA, nullptr, HID_, HID_, HID_, 0, 0);
    gemm(hA,   p.qw,  p.qb,  nullptr, qh, HID_, HID_, DVDIM, 0, 0);
    gemm(hA,   p.kw,  p.kb,  nullptr, kh, HID_, HID_, DVDIM, 0, 0);
    gemm(hA,   p.vw,  p.vb,  nullptr, vh, HID_, HID_, DVDIM, 0, 0);
    attn_kernel<<<dim3(NSEQ / 128, NBATCH), 256, 0, stream>>>(qh, kh, vh, oattn);
    gemm(oattn, p.ow, p.ob, sabout, nullptr, DVDIM, DVDIM, DVDIM, 1, 1);
    spline_rqs<<<(NTOK * E2_) / 256, 256, 0, stream>>>(ubuf, sabout, ubuf, ldout);
  };

  slogdet_init<<<1, 64, 0, stream>>>(convW, ldout);
  conv_split<<<(NTOK * E_) / 256, 256, 0, stream>>>(xin, convW, lower, upper);
  coupling(lower, upper, blk[0]);   // f1 transforms upper, conditioned on lower
  coupling(upper, lower, blk[1]);   // f2 transforms lower, conditioned on new upper
  concat_out<<<(NTOK * E_) / 256, 256, 0, stream>>>(lower, upper, outT);
}
